// FSM_62380105007688
// MI455X (gfx1250) — matrix operand order repaired and checked by reading, not compiled, not run on hardware
//
#include <hip/hip_runtime.h>
#include <hip/hip_bf16.h>

typedef __attribute__((ext_vector_type(16))) __bf16 v16bf;
typedef __attribute__((ext_vector_type(8)))  __bf16 v8bf;
typedef __attribute__((ext_vector_type(4)))  __bf16 v4bf;
typedef __attribute__((ext_vector_type(2)))  __bf16 v2bf;
typedef __attribute__((ext_vector_type(8)))  float  v8f;
typedef unsigned int u32;
typedef __attribute__((ext_vector_type(4))) unsigned int uint32x4;
typedef __attribute__((ext_vector_type(8))) int          int32x8;
typedef __attribute__((ext_vector_type(4))) int          int32x4;

#if defined(__has_builtin)
#if __has_builtin(__builtin_amdgcn_tensor_load_to_lds)
#define HAVE_TDM 1
#endif
#endif
#if __has_include(<hip/amd_detail/amd_gfx1250_TDM.h>)
#define TDM_6ARG 1
#endif

// ---------------------------------------------------------------------------
// WMMA helper (CDNA5 wave32, 16x16x32 bf16 -> f32 accumulate)
// ---------------------------------------------------------------------------
__device__ __forceinline__ v8f wmma_bf16(v16bf a, v16bf b, v8f c) {
  return __builtin_amdgcn_wmma_f32_16x16x32_bf16(false, a, false, b, (short)0, c,
                                                 false, false);
}

// 16-bit B matrix 32x16 (KxN), lane L holds column N=L&15.
// k order used here (not AMD's table): lanes 0-15: K=e ; lanes 16-31: K=16+e
__device__ __forceinline__ int b_k(int lane, int e) {
  return ((lane < 16) ? 0 : 16) + e;
}

#define HW   4096
#define CIN  256
#define C8   32
#define C16  16
#define NB   4
#define TSTRIDE 40   // padded channel count: 80B row stride, 16B aligned, bank-spread

// ---------------------------------------------------------------------------
// TDM staging: 2-D tile = 32 channel-rows x 192 contiguous bf16 (3 image rows)
// LDS layout after load: tileA[c*192 + r*64 + col]
// ---------------------------------------------------------------------------
__device__ __forceinline__ void stage_tile(__bf16* lds, const __bf16* g,
                                           int tid, int nthreads) {
#if defined(HAVE_TDM)
  if (tid < 32) {  // one wave issues the DMA (EXEC-independent instruction)
    unsigned long long ga = (unsigned long long)(const void*)g;
    u32 lds_off = (u32)(unsigned long long)(void*)lds;  // low 32 bits = LDS addr
    uint32x4 g0;
    int32x8  g1;
    int32x4  g2 = {0, 0, 0, 0}, g3 = {0, 0, 0, 0};
    g0[0] = 1u;                                   // count=1 (valid), user mode
    g0[1] = lds_off;                              // lds_addr
    g0[2] = (u32)(ga & 0xffffffffu);              // global_addr[31:0]
    g0[3] = (u32)((ga >> 32) & 0x01ffffffu) | 0x80000000u;  // addr[56:32]|type=2
    g1[0] = (int)(1u << 16);                      // data_size=1 (2B), mask=0
    g1[1] = (int)(192u << 16);                    // tensor_dim0[15:0]=192
    g1[2] = (int)(32u << 16);                     // tensor_dim1[15:0]=32
    g1[3] = (int)(192u << 16);                    // tile_dim0=192
    g1[4] = (int)32;                              // tile_dim1=32, tile_dim2=0
    g1[5] = (int)4096;                            // tensor_dim0_stride=4096
    g1[6] = 0;
    g1[7] = 0;
#if defined(TDM_6ARG)
    int32x8 g4 = {0, 0, 0, 0, 0, 0, 0, 0};
    __builtin_amdgcn_tensor_load_to_lds(g0, g1, g2, g3, g4, 0);
#else
    __builtin_amdgcn_tensor_load_to_lds(g0, g1, g2, g3, 0);
#endif
    __builtin_amdgcn_s_wait_tensorcnt(0);
  }
#else
  for (int i = tid; i < (C8 * 192) / 8; i += nthreads) {
    int c = i / 24, w = i % 24;
    *(uint4*)(lds + c * 192 + w * 8) = *(const uint4*)(g + (size_t)c * HW + w * 8);
  }
#endif
}

// channel-major [32][192] -> channel-innermost [3][64][TSTRIDE]
__device__ __forceinline__ void transpose_tile(const __bf16* tileA, __bf16* tileT,
                                               int tid, int nthreads) {
  for (int ch = tid; ch < C8 * 24; ch += nthreads) {  // 768 8-col chunks
    int c = ch / 24, q = ch % 24;
    int r = q >> 3, colg = (q & 7) << 3;
    v8bf v = *(const v8bf*)(tileA + c * 192 + r * 64 + colg);
#pragma unroll
    for (int e = 0; e < 8; ++e)
      tileT[(r * 64 + colg + e) * TSTRIDE + c] = v[e];
  }
}

// A fragment: lane (M=lane&15) needs channels {base..base+15}, base=0/16 by lane half
// at fixed (r,col): two contiguous 16B LDS loads.
__device__ __forceinline__ v16bf load_a_frag(const __bf16* tileT, int r, int colc,
                                             int lane, bool ok) {
  const __bf16* p = tileT + (r * 64 + colc) * TSTRIDE + ((lane < 16) ? 0 : 16);
  v8bf lo = *(const v8bf*)p;
  v8bf hi = *(const v8bf*)(p + 8);
  v16bf a;
#pragma unroll
  for (int e = 0; e < 8; ++e) { a[e] = lo[e]; a[8 + e] = hi[e]; }
  if (!ok) {
#pragma unroll
    for (int e = 0; e < 16; ++e) a[e] = (__bf16)0.0f;
  }
  return a;
}

// ---------------------------------------------------------------------------
// conversion kernels
// ---------------------------------------------------------------------------
__global__ void k_cvt_x(const float* __restrict__ x, __bf16* __restrict__ xbf) {
  int i = blockIdx.x * blockDim.x + threadIdx.x;  // one per 4 elements
  float4 v = ((const float4*)x)[i];
  v4bf o = {(__bf16)v.x, (__bf16)v.y, (__bf16)v.z, (__bf16)v.w};
  *(v4bf*)(xbf + (size_t)i * 4) = o;
}

// conv1_w [32,256,3,3] -> fragment layout W1F[(t*8+cs)*2+nt][lane][e]
__global__ void k_cvt_w1f(const float* __restrict__ w, __bf16* __restrict__ wf) {
  int i = blockIdx.x * blockDim.x + threadIdx.x;  // 0..4607 (144 frags * 32 lanes)
  if (i >= 144 * 32) return;
  int lane = i & 31, f = i >> 5;
  int t = f / 16, rem = f % 16, cs = rem >> 1, nt = rem & 1;
  int oc = nt * 16 + (lane & 15);
  const float* src = w + (size_t)oc * (CIN * 9) + (cs * 32 + ((lane < 16) ? 0 : 16)) * 9 + t;
  v16bf o;
#pragma unroll
  for (int e = 0; e < 16; ++e) o[e] = (__bf16)src[e * 9];
  *(v16bf*)(wf + (size_t)i * 16) = o;
}

// conv2_w [256,32,3,3] -> fragment layout W2F[t*16+nt][lane][e]
__global__ void k_cvt_w2f(const float* __restrict__ w, __bf16* __restrict__ wf) {
  int i = blockIdx.x * blockDim.x + threadIdx.x;  // 0..4607
  if (i >= 144 * 32) return;
  int lane = i & 31, f = i >> 5;
  int t = f / 16, nt = f % 16;
  int oc = nt * 16 + (lane & 15);
  const float* src = w + (size_t)oc * (C8 * 9) + ((lane < 16) ? 0 : 16) * 9 + t;
  v16bf o;
#pragma unroll
  for (int e = 0; e < 16; ++e) o[e] = (__bf16)src[e * 9];
  *(v16bf*)(wf + (size_t)i * 16) = o;
}

// ---------------------------------------------------------------------------
// conv1: 3x3, 256->32.  256 blocks (b,h) x 8 waves: mt=wv&3, nt=wv>>2.
// Per 32-ch k-block: TDM stage -> LDS transpose -> 9 taps (b128 A frags).
// ---------------------------------------------------------------------------
__global__ void __launch_bounds__(256) k_conv1(
    const __bf16* __restrict__ xbf, const __bf16* __restrict__ w1f,
    const float* __restrict__ bias, float* __restrict__ out) {
  __shared__ __bf16 tileA[C8 * 192];            // 12 KB
  __shared__ __bf16 tileT[3 * 64 * TSTRIDE];    // 15 KB
  int lane = threadIdx.x & 31;
  int wv   = threadIdx.x >> 5;
  int b = blockIdx.x >> 6;
  int h = blockIdx.x & 63;
  int w0 = (wv & 3) << 4;
  int nt = wv >> 2;
  int M = lane & 15, N = lane & 15;
  v8f acc = {};
  for (int cs = 0; cs < 8; ++cs) {
    __syncthreads();
    stage_tile(tileA, xbf + ((size_t)(b * CIN + cs * 32)) * HW + (h - 1) * 64,
               threadIdx.x, 256);
    __syncthreads();
    transpose_tile(tileA, tileT, threadIdx.x, 256);
    __syncthreads();
#pragma unroll
    for (int t = 0; t < 9; ++t) {
      int kh = t / 3 - 1, kw = t % 3 - 1;
      int hh = h + kh;
      int col = w0 + M + kw;
      bool ok = (hh >= 0) && (hh < 64) && (col >= 0) && (col < 64);
      int colc = col < 0 ? 0 : (col > 63 ? 63 : col);
      v16bf a = load_a_frag(tileT, kh + 1, colc, lane, ok);
      v16bf bb = *(const v16bf*)(w1f + (size_t)((((t * 8 + cs) * 2 + nt) * 32) + lane) * 16);
      acc = wmma_bf16(a, bb, acc);
    }
  }
  int oc = nt * 16 + N;
  float bv = bias[oc];
  float4 lo = {acc[0] + bv, acc[1] + bv, acc[2] + bv, acc[3] + bv};
  float4 hi = {acc[4] + bv, acc[5] + bv, acc[6] + bv, acc[7] + bv};
  float* o = out + ((size_t)b * C8 + oc) * HW + h * 64 + w0 + ((lane < 16) ? 0 : 8);
  *(float4*)o = lo;
  *(float4*)(o + 4) = hi;
}

// ---------------------------------------------------------------------------
// group-norm stats -> per-channel scale/shift
// ---------------------------------------------------------------------------
__global__ void k_gnstats(const float* __restrict__ src, const float* __restrict__ gw,
                          const float* __restrict__ gb, float* __restrict__ ss,
                          int C, int cpg) {
  int groups = C / cpg;
  int b = blockIdx.x / groups, g = blockIdx.x % groups;
  const float* base = src + ((size_t)b * C + g * cpg) * HW;
  int n = cpg * HW;
  float s = 0.f, sq = 0.f;
  for (int i = threadIdx.x; i < n / 4; i += blockDim.x) {
    float4 v = ((const float4*)base)[i];
    s += v.x + v.y + v.z + v.w;
    sq += v.x * v.x + v.y * v.y + v.z * v.z + v.w * v.w;
  }
  __shared__ float sh_s[256], sh_q[256];
  sh_s[threadIdx.x] = s; sh_q[threadIdx.x] = sq;
  __syncthreads();
  for (int st = blockDim.x >> 1; st > 0; st >>= 1) {
    if ((int)threadIdx.x < st) {
      sh_s[threadIdx.x] += sh_s[threadIdx.x + st];
      sh_q[threadIdx.x] += sh_q[threadIdx.x + st];
    }
    __syncthreads();
  }
  if (threadIdx.x == 0) {
    float mean = sh_s[0] / n;
    float var  = sh_q[0] / n - mean * mean;
    float rstd = rsqrtf(var + 1e-5f);
    for (int c = g * cpg; c < (g + 1) * cpg; ++c) {
      float sc = gw[c] * rstd;
      ss[((size_t)b * C + c) * 2]     = sc;
      ss[((size_t)b * C + c) * 2 + 1] = gb[c] - mean * sc;
    }
  }
}

// ---------------------------------------------------------------------------
// GN1 apply + ReLU + 1x1 projections theta/phi/g; 2 pixels/thread (b64 I/O)
// ---------------------------------------------------------------------------
__global__ void k_gn1_tpg(const float* __restrict__ c1out, const float* __restrict__ ss,
                          const float* __restrict__ thw, const float* __restrict__ phw,
                          const float* __restrict__ ggw, float* __restrict__ ip,
                          __bf16* __restrict__ Tb, __bf16* __restrict__ Pb,
                          __bf16* __restrict__ Gb) {
  __shared__ float wth[C16 * C8], wph[C16 * C8], wg[C16 * C8];
  for (int i = threadIdx.x; i < C16 * C8; i += blockDim.x) {
    wth[i] = thw[i]; wph[i] = phw[i]; wg[i] = ggw[i];
  }
  __syncthreads();
  int p = blockIdx.x * blockDim.x + threadIdx.x;  // 0..8191, 2 px each
  int b = p >> 11, hw = (p & 2047) * 2;
  float2 v[C8];
#pragma unroll 4
  for (int c = 0; c < C8; ++c) {
    size_t idx = ((size_t)b * C8 + c) * HW + hw;
    float sc = ss[((size_t)b * C8 + c) * 2], sh = ss[((size_t)b * C8 + c) * 2 + 1];
    float2 xv = *(const float2*)(c1out + idx);
    float2 y = {xv.x * sc + sh, xv.y * sc + sh};
    y.x = y.x > 0.f ? y.x : 0.f;
    y.y = y.y > 0.f ? y.y : 0.f;
    v[c] = y;
    *(float2*)(ip + idx) = y;
  }
  for (int o = 0; o < C16; ++o) {
    float2 st = {0.f, 0.f}, sp = {0.f, 0.f}, sg = {0.f, 0.f};
#pragma unroll 8
    for (int c = 0; c < C8; ++c) {
      float wt = wth[o * C8 + c], wp = wph[o * C8 + c], wgv = wg[o * C8 + c];
      st.x += wt * v[c].x;  st.y += wt * v[c].y;
      sp.x += wp * v[c].x;  sp.y += wp * v[c].y;
      sg.x += wgv * v[c].x; sg.y += wgv * v[c].y;
    }
    size_t idx = (size_t)b * (C16 * HW) + o * HW + hw;
    *(v2bf*)(Tb + idx) = (v2bf){(__bf16)st.x, (__bf16)st.y};
    *(v2bf*)(Pb + idx) = (v2bf){(__bf16)sp.x, (__bf16)sp.y};
    *(v2bf*)(Gb + idx) = (v2bf){(__bf16)sg.x, (__bf16)sg.y};
  }
}

// ---------------------------------------------------------------------------
// S = phi_view[16,4096] @ g_view[4096,16] / N  -> stored TRANSPOSED (STt[n][i])
// ---------------------------------------------------------------------------
__global__ void k_S(const __bf16* __restrict__ Pb, const __bf16* __restrict__ Gb,
                    __bf16* __restrict__ STt) {
  int b = blockIdx.x;
  int lane = threadIdx.x & 31;
  const __bf16* P = Pb + (size_t)b * (C16 * HW);
  const __bf16* G = Gb + (size_t)b * (C16 * HW);
  int M = lane & 15, N = lane & 15;
  int abase = (lane < 16) ? 0 : 16;
  int bbase = (lane < 16) ? 0 : 16;
  v8f acc = {};
  for (int ks = 0; ks < 128; ++ks) {
    int m0 = ks * 32;
    v8bf lo = *(const v8bf*)(P + (size_t)M * HW + m0 + abase);
    v8bf hi = *(const v8bf*)(P + (size_t)M * HW + m0 + 8 + abase);
    v16bf a, bb;
#pragma unroll
    for (int e = 0; e < 8; ++e) { a[e] = lo[e]; a[8 + e] = hi[e]; }
#pragma unroll
    for (int e = 0; e < 16; ++e) bb[e] = G[(size_t)(m0 + bbase + e) * C16 + N];
    acc = wmma_bf16(a, bb, acc);
  }
  const float inv = 1.0f / (float)HW;
#pragma unroll
  for (int r = 0; r < 8; ++r) {
    int i = r + ((lane < 16) ? 0 : 8);
    STt[(size_t)b * 256 + N * 16 + i] = (__bf16)(acc[r] * inv);  // transposed
  }
}

// ---------------------------------------------------------------------------
// Y = theta_view[4096,16] @ S[16,16];  32B row loads, K padded with zeros
// ---------------------------------------------------------------------------
__global__ void k_Y(const __bf16* __restrict__ Tb, const __bf16* __restrict__ STt,
                    float* __restrict__ Yf) {
  int lane = threadIdx.x & 31;
  int wid  = blockIdx.x * (blockDim.x >> 5) + (threadIdx.x >> 5);  // 0..1023
  int b = wid >> 8, tile = wid & 255;
  int n0 = tile * 16;
  int M = lane & 15, N = lane & 15;
  v16bf row  = *(const v16bf*)(Tb + (size_t)b * (C16 * HW) + (size_t)(n0 + M) * 16);
  v16bf srow = *(const v16bf*)(STt + (size_t)b * 256 + N * 16);
  bool lolane = lane < 16;
  v16bf a, bb;
#pragma unroll
  for (int e = 0; e < 8; ++e) {
    a[e]     = lolane ? row[e] : row[8 + e];
    a[8 + e] = (__bf16)0.0f;
  }
#pragma unroll
  for (int e = 0; e < 16; ++e) bb[e] = (e < 8) ? (lolane ? srow[e] : srow[8 + e]) : (__bf16)0.0f;
  v8f acc = {};
  acc = wmma_bf16(a, bb, acc);
#pragma unroll
  for (int r = 0; r < 8; ++r) {
    int m = r + (lolane ? 0 : 8);
    Yf[(size_t)b * (C16 * HW) + (size_t)(n0 + m) * 16 + N] = acc[r];
  }
}

// ---------------------------------------------------------------------------
// z = out_w @ y_view + ip; 4 pixels/thread (float4 / v4bf I/O)
// ---------------------------------------------------------------------------
__global__ void k_z(const float* __restrict__ Yf, const float* __restrict__ outw,
                    const float* __restrict__ ip, float* __restrict__ zf,
                    __bf16* __restrict__ zb) {
  __shared__ float w[C8 * C16];
  for (int i = threadIdx.x; i < C8 * C16; i += blockDim.x) w[i] = outw[i];
  __syncthreads();
  int i = blockIdx.x * blockDim.x + threadIdx.x;  // 0..131071, 4 px each
  int base = i * 4;
  int b = base >> 17;
  int rem = base & 131071;
  int o2 = rem >> 12, hw = rem & 4095;
  const float* Y = Yf + (size_t)b * (C16 * HW);
  float4 s = *(const float4*)(ip + base);
#pragma unroll
  for (int o = 0; o < C16; ++o) {
    float wv = w[o2 * C16 + o];
    float4 yv = *(const float4*)(Y + o * HW + hw);
    s.x += wv * yv.x; s.y += wv * yv.y; s.z += wv * yv.z; s.w += wv * yv.w;
  }
  *(float4*)(zf + base) = s;
  *(v4bf*)(zb + base) = (v4bf){(__bf16)s.x, (__bf16)s.y, (__bf16)s.z, (__bf16)s.w};
}

// ---------------------------------------------------------------------------
// conv2: 3x3, 32->256.  256 blocks (b,h) x 16 waves: mt=wv&3, nq=wv>>2.
// TDM stage once -> transpose -> 9 taps x 4 N-tiles = 36 WMMA/wave.
// ---------------------------------------------------------------------------
__global__ void __launch_bounds__(512) k_conv2(
    const __bf16* __restrict__ zb, const __bf16* __restrict__ w2f,
    const float* __restrict__ bias, float* __restrict__ out) {
  __shared__ __bf16 tileA[C8 * 192];
  __shared__ __bf16 tileT[3 * 64 * TSTRIDE];
  int lane = threadIdx.x & 31;
  int wv   = threadIdx.x >> 5;
  int b = blockIdx.x >> 6;
  int h = blockIdx.x & 63;
  int w0 = (wv & 3) << 4;
  int nq = wv >> 2;
  int M = lane & 15, N = lane & 15;
  stage_tile(tileA, zb + (size_t)b * C8 * HW + (h - 1) * 64, threadIdx.x, 512);
  __syncthreads();
  transpose_tile(tileA, tileT, threadIdx.x, 512);
  __syncthreads();
  v8f acc0 = {}, acc1 = {}, acc2 = {}, acc3 = {};
#pragma unroll
  for (int t = 0; t < 9; ++t) {
    int kh = t / 3 - 1, kw = t % 3 - 1;
    int hh = h + kh;
    int col = w0 + M + kw;
    bool ok = (hh >= 0) && (hh < 64) && (col >= 0) && (col < 64);
    int colc = col < 0 ? 0 : (col > 63 ? 63 : col);
    v16bf a = load_a_frag(tileT, kh + 1, colc, lane, ok);
    const __bf16* wt = w2f + (size_t)((t * 16 + nq * 4) * 32 + lane) * 16;
    v16bf b0 = *(const v16bf*)(wt);
    v16bf b1 = *(const v16bf*)(wt + 512);
    v16bf b2 = *(const v16bf*)(wt + 1024);
    v16bf b3 = *(const v16bf*)(wt + 1536);
    acc0 = wmma_bf16(a, b0, acc0);
    acc1 = wmma_bf16(a, b1, acc1);
    acc2 = wmma_bf16(a, b2, acc2);
    acc3 = wmma_bf16(a, b3, acc3);
  }
  int hwb = h * 64 + w0 + ((lane < 16) ? 0 : 8);
  v8f* accs[4] = {&acc0, &acc1, &acc2, &acc3};
#pragma unroll
  for (int j = 0; j < 4; ++j) {
    int oc = nq * 64 + j * 16 + N;
    float bv = bias[oc];
    v8f& a8 = *accs[j];
    float4 lo = {a8[0] + bv, a8[1] + bv, a8[2] + bv, a8[3] + bv};
    float4 hi = {a8[4] + bv, a8[5] + bv, a8[6] + bv, a8[7] + bv};
    float* o = out + ((size_t)b * CIN + oc) * HW + hwb;
    *(float4*)o = lo;
    *(float4*)(o + 4) = hi;
  }
}

// ---------------------------------------------------------------------------
// final: GN2 apply + ReLU + outer residual; 4 elements/thread
// ---------------------------------------------------------------------------
__global__ void k_final(const float* __restrict__ c2out, const float* __restrict__ ss,
                        const float* __restrict__ x, float* __restrict__ out) {
  int i = blockIdx.x * blockDim.x + threadIdx.x;  // 0..1048575, 4 elems each
  int base = i * 4;
  int b = base >> 20;
  int c = (base >> 12) & 255;
  float sc = ss[((size_t)b * CIN + c) * 2], sh = ss[((size_t)b * CIN + c) * 2 + 1];
  float4 v = *(const float4*)(c2out + base);
  float4 xr = *(const float4*)(x + base);
  float4 y;
  y.x = v.x * sc + sh; y.y = v.y * sc + sh;
  y.z = v.z * sc + sh; y.w = v.w * sc + sh;
  y.x = (y.x > 0.f ? y.x : 0.f) + xr.x;
  y.y = (y.y > 0.f ? y.y : 0.f) + xr.y;
  y.z = (y.z > 0.f ? y.z : 0.f) + xr.z;
  y.w = (y.w > 0.f ? y.w : 0.f) + xr.w;
  *(float4*)(out + base) = y;
}

// ---------------------------------------------------------------------------
// launch
// ---------------------------------------------------------------------------
extern "C" void kernel_launch(void* const* d_in, const int* in_sizes, int n_in,
                              void* d_out, int out_size, void* d_ws, size_t ws_size,
                              hipStream_t stream) {
  const float* x       = (const float*)d_in[0];
  const float* conv1_w = (const float*)d_in[1];
  const float* conv1_b = (const float*)d_in[2];
  const float* gn1_w   = (const float*)d_in[3];
  const float* gn1_b   = (const float*)d_in[4];
  const float* theta_w = (const float*)d_in[5];
  const float* phi_w   = (const float*)d_in[6];
  const float* g_w     = (const float*)d_in[7];
  const float* out_w   = (const float*)d_in[8];
  const float* conv2_w = (const float*)d_in[9];
  const float* conv2_b = (const float*)d_in[10];
  const float* gn2_w   = (const float*)d_in[11];
  const float* gn2_b   = (const float*)d_in[12];
  float* out = (float*)d_out;

  char* ws = (char*)d_ws;
  size_t off = 0;
  auto alloc = [&](size_t bytes) -> char* {
    char* p = ws + off;
    off = (off + bytes + 255) & ~(size_t)255;
    return p;
  };
  (void)alloc(256);  // guard region: halo DMA before row 0 lands here
  __bf16* xbf   = (__bf16*)alloc((size_t)NB * CIN * HW * 2);
  __bf16* w1f   = (__bf16*)alloc((size_t)C8 * CIN * 9 * 2);
  __bf16* w2f   = (__bf16*)alloc((size_t)CIN * C8 * 9 * 2);
  float*  c1out = (float*) alloc((size_t)NB * C8 * HW * 4);
  float*  gn1ss = (float*) alloc((size_t)NB * C8 * 2 * 4);
  float*  ip    = (float*) alloc((size_t)NB * C8 * HW * 4);
  __bf16* Tb    = (__bf16*)alloc((size_t)NB * C16 * HW * 2);
  __bf16* Pb    = (__bf16*)alloc((size_t)NB * C16 * HW * 2);
  __bf16* Gb    = (__bf16*)alloc((size_t)NB * C16 * HW * 2);
  __bf16* STt   = (__bf16*)alloc((size_t)NB * 256 * 2);
  float*  Yf    = (float*) alloc((size_t)NB * C16 * HW * 4);
  float*  zf    = (float*) alloc((size_t)NB * C8 * HW * 4);
  __bf16* zbuf  = (__bf16*)alloc((size_t)NB * C8 * HW * 2);
  float*  c2out = (float*) alloc((size_t)NB * CIN * HW * 4);
  float*  gn2ss = (float*) alloc((size_t)NB * CIN * 2 * 4);
  (void)alloc(256);  // trailing guard for halo DMA past last row
  (void)ws_size; (void)n_in; (void)in_sizes; (void)out_size;

  int nx = NB * CIN * HW;  // 4,194,304
  k_cvt_x  <<<nx / 4 / 256, 256, 0, stream>>>(x, xbf);
  k_cvt_w1f<<<(144 * 32 + 255) / 256, 256, 0, stream>>>(conv1_w, w1f);
  k_cvt_w2f<<<(144 * 32 + 255) / 256, 256, 0, stream>>>(conv2_w, w2f);

  k_conv1<<<NB * 64, 256, 0, stream>>>(xbf, w1f, conv1_b, c1out);
  k_gnstats<<<NB * C8, 256, 0, stream>>>(c1out, gn1_w, gn1_b, gn1ss, C8, 1);
  k_gn1_tpg<<<(NB * HW / 2) / 256, 256, 0, stream>>>(c1out, gn1ss, theta_w, phi_w,
                                                     g_w, ip, Tb, Pb, Gb);
  k_S<<<NB, 32, 0, stream>>>(Pb, Gb, STt);
  k_Y<<<256, 128, 0, stream>>>(Tb, STt, Yf);
  k_z<<<(NB * C8 * HW / 4) / 256, 256, 0, stream>>>(Yf, out_w, ip, zf, zbuf);
  k_conv2<<<NB * 64, 512, 0, stream>>>(zbuf, w2f, conv2_b, c2out);
  k_gnstats<<<NB * 32, 256, 0, stream>>>(c2out, gn2_w, gn2_b, gn2ss, CIN, 8);
  k_final<<<nx / 4 / 256, 256, 0, stream>>>(c2out, gn2ss, x, out);
}
